// QuantumLayer_50053548867619
// MI455X (gfx1250) — matrix operand order repaired and checked by reading, not compiled, not run on hardware
//
#include <hip/hip_runtime.h>
#include <math.h>

typedef __attribute__((ext_vector_type(16))) _Float16 v16h;
typedef __attribute__((ext_vector_type(8)))  float    v8f;

#define INPUT_DIM 512
#define NQ        8
#define BATCH     8192
#define NLAYERS   3

// ---------------------------------------------------------------------------
// Kernel 1: proj = x @ W.T + b ; angles = sigmoid(proj*scale+bias)*pi
// One wave per 16-row tile of x; N=8 padded to 16; K in steps of 32 via
// v_wmma_f32_16x16x32_f16 (16 WMMAs per wave).
// ---------------------------------------------------------------------------
__global__ __launch_bounds__(256) void qnn_gemm_angles(
    const float* __restrict__ x, const float* __restrict__ W,
    const float* __restrict__ bvec, const float* __restrict__ scale,
    const float* __restrict__ bias, float* __restrict__ angles)
{
  const int lane = threadIdx.x & 31;
  const int wave = threadIdx.x >> 5;
  const int tile = blockIdx.x * 8 + wave;     // 512 tiles of 16 batch rows
  const int h    = lane >> 4;                 // lane half (0/1)
  const int m    = lane & 15;                 // A row within tile
  const int n    = lane & 15;                 // C/D column (qubit), padded
  const int nc   = (n < NQ) ? n : 0;          // clamped row of W (keeps EXEC full)
  const float nmask = (n < NQ) ? 1.0f : 0.0f;

  const float* xrow = x + (size_t)(tile * 16 + m) * INPUT_DIM;
  const float* wrow = W + (size_t)nc * INPUT_DIM;

  v8f acc = {};
  for (int kk = 0; kk < INPUT_DIM; kk += 32) {
    v16h aF, bF;
    // A 16x32 f16 layout: lane half h holds K = kk+8h+j (j<8) and kk+16+8h+(j-8)
    #pragma unroll
    for (int j = 0; j < 8; ++j) {
      aF[j]     = (_Float16)xrow[kk +      8 * h + j];
      aF[j + 8] = (_Float16)xrow[kk + 16 + 8 * h + j];
    }
    // B 32x16 f16 layout: lane n = col, half h holds K = kk+8h+j (j<8) and kk+16+8h+(j-8)
    #pragma unroll
    for (int j = 0; j < 8; ++j) {
      bF[j]     = (_Float16)(wrow[kk +      8 * h + j] * nmask);
      bF[j + 8] = (_Float16)(wrow[kk + 16 + 8 * h + j] * nmask);
    }

    acc = __builtin_amdgcn_wmma_f32_16x16x32_f16(false, aF, false, bF,
                                                 (short)0, acc, false, false);
  }

  const float PI = 3.14159265358979323846f;
  const float bb = bvec[nc], sc = scale[nc], bi = bias[nc];
  if (n < NQ) {
    #pragma unroll
    for (int r = 0; r < 8; ++r) {
      const int   mg   = tile * 16 + r + 8 * h;   // C/D: VGPR r -> row r+8h
      const float proj = acc[r] + bb;
      const float t    = proj * sc + bi;
      angles[(size_t)mg * NQ + n] = PI / (1.0f + __expf(-t));
    }
  }
}

// ---------------------------------------------------------------------------
// Kernel 2: 8-qubit state-vector simulation, one batch element per wave32.
// Amplitude index i in [0,256): lane = i[7:3], register r = i[2:0].
// Qubit w acts on bit (7-w) of i (axis 1 of the reshape is the MSB).
// ---------------------------------------------------------------------------
__device__ __forceinline__ void apply1q(
    float sre[8], float sim_[8], int bit, int lane,
    float u00r, float u00i, float u01r, float u01i,
    float u10r, float u10i, float u11r, float u11i)
{
  if (bit < 3) {                                  // register-bit pairing
    const int mask = 1 << bit;
    #pragma unroll
    for (int r = 0; r < 8; ++r) {
      if (r & mask) continue;
      const int r1 = r | mask;
      const float a0r = sre[r],  a0i = sim_[r];
      const float a1r = sre[r1], a1i = sim_[r1];
      sre[r]   = u00r*a0r - u00i*a0i + u01r*a1r - u01i*a1i;
      sim_[r]  = u00r*a0i + u00i*a0r + u01r*a1i + u01i*a1r;
      sre[r1]  = u10r*a0r - u10i*a0i + u11r*a1r - u11i*a1i;
      sim_[r1] = u10r*a0i + u10i*a0r + u11r*a1i + u11i*a1r;
    }
  } else {                                        // lane-bit pairing: shfl_xor
    const int  lm = 1 << (bit - 3);
    const bool hi = (lane & lm) != 0;
    #pragma unroll
    for (int r = 0; r < 8; ++r) {
      const float pre = __shfl_xor(sre[r],  lm, 32);
      const float pim = __shfl_xor(sim_[r], lm, 32);
      const float a0r = hi ? pre     : sre[r];
      const float a0i = hi ? pim     : sim_[r];
      const float a1r = hi ? sre[r]  : pre;
      const float a1i = hi ? sim_[r] : pim;
      const float m0r = hi ? u10r : u00r, m0i = hi ? u10i : u00i;
      const float m1r = hi ? u11r : u01r, m1i = hi ? u11i : u01i;
      sre[r]  = m0r*a0r - m0i*a0i + m1r*a1r - m1i*a1i;
      sim_[r] = m0r*a0i + m0i*a0r + m1r*a1i + m1i*a1r;
    }
  }
}

__device__ __forceinline__ void applyCNOT(
    float sre[8], float sim_[8], int bc, int bt, int lane)
{
  float tre[8], tim[8];
  if (bt < 3) {                                   // target is a register bit
    const int tm = 1 << bt;
    #pragma unroll
    for (int r = 0; r < 8; ++r) {
      const int  i    = (lane << 3) | r;
      const int  rp   = r ^ tm;
      const bool ctrl = ((i >> bc) & 1) != 0;
      tre[r] = ctrl ? sre[rp]  : sre[r];
      tim[r] = ctrl ? sim_[rp] : sim_[r];
    }
  } else {                                        // target is a lane bit
    const int lm = 1 << (bt - 3);
    #pragma unroll
    for (int r = 0; r < 8; ++r) {
      const float pre = __shfl_xor(sre[r],  lm, 32);
      const float pim = __shfl_xor(sim_[r], lm, 32);
      const int   i   = (lane << 3) | r;
      const bool ctrl = ((i >> bc) & 1) != 0;
      tre[r] = ctrl ? pre : sre[r];
      tim[r] = ctrl ? pim : sim_[r];
    }
  }
  #pragma unroll
  for (int r = 0; r < 8; ++r) { sre[r] = tre[r]; sim_[r] = tim[r]; }
}

__global__ __launch_bounds__(256) void qnn_sim(
    const float* __restrict__ angles, const float* __restrict__ qw,
    float* __restrict__ out)
{
  const int lane = threadIdx.x & 31;
  const int wave = threadIdx.x >> 5;
  const int b    = blockIdx.x * 8 + wave;

  float a[NQ];
  #pragma unroll
  for (int w = 0; w < NQ; ++w) a[w] = angles[(size_t)b * NQ + w];

  float sre[8], sim_[8];
  #pragma unroll
  for (int r = 0; r < 8; ++r) { sre[r] = 0.f; sim_[r] = 0.f; }
  sre[0] = (lane == 0) ? 1.0f : 0.0f;             // |0...0>

  // AngleEmbedding: RY(a[w]) on wire w  (real 2x2: [[c,-s],[s,c]])
  #pragma unroll
  for (int w = 0; w < NQ; ++w) {
    const float c = cosf(a[w] * 0.5f), s = sinf(a[w] * 0.5f);
    apply1q(sre, sim_, 7 - w, lane, c, 0.f, -s, 0.f, s, 0.f, c, 0.f);
  }

  // StronglyEntanglingLayers
  for (int l = 0; l < NLAYERS; ++l) {
    #pragma unroll
    for (int w = 0; w < NQ; ++w) {
      const float phi = qw[(l * NQ + w) * 3 + 0];
      const float th  = qw[(l * NQ + w) * 3 + 1];
      const float om  = qw[(l * NQ + w) * 3 + 2];
      const float c  = cosf(th * 0.5f), s = sinf(th * 0.5f);
      const float pp = (phi + om) * 0.5f, pm = (phi - om) * 0.5f;
      const float cp = cosf(pp), sp = sinf(pp);
      const float cm = cosf(pm), sm = sinf(pm);
      // u00 = e^{-i pp} c ; u01 = -e^{+i pm} s ; u10 = e^{-i pm} s ; u11 = e^{+i pp} c
      apply1q(sre, sim_, 7 - w, lane,
              cp * c, -sp * c,   -cm * s, -sm * s,
              cm * s, -sm * s,    cp * c,  sp * c);
    }
    const int rr = (l % (NQ - 1)) + 1;
    for (int w = 0; w < NQ; ++w) {
      const int t = (w + rr) % NQ;
      applyCNOT(sre, sim_, 7 - w, 7 - t, lane);
    }
  }

  // <Z_w> = sum_i (1 - 2*bit_{7-w}(i)) |amp_i|^2 , wave-reduced
  float p[8];
  #pragma unroll
  for (int r = 0; r < 8; ++r) p[r] = sre[r] * sre[r] + sim_[r] * sim_[r];

  float z[NQ];
  #pragma unroll
  for (int w = 0; w < NQ; ++w) {
    const int bit = 7 - w;
    float acc = 0.f;
    #pragma unroll
    for (int r = 0; r < 8; ++r) {
      const int i = (lane << 3) | r;
      acc += (((i >> bit) & 1) != 0) ? -p[r] : p[r];
    }
    #pragma unroll
    for (int off = 16; off >= 1; off >>= 1)
      acc += __shfl_xor(acc, off, 32);
    z[w] = acc;
  }

  if (lane == 0) {
    #pragma unroll
    for (int w = 0; w < NQ; ++w) out[(size_t)b * NQ + w] = z[w];
  }
}

// ---------------------------------------------------------------------------
extern "C" void kernel_launch(void* const* d_in, const int* in_sizes, int n_in,
                              void* d_out, int out_size, void* d_ws, size_t ws_size,
                              hipStream_t stream) {
  const float* x     = (const float*)d_in[0];   // (8192, 512)
  const float* W     = (const float*)d_in[1];   // (8, 512)
  const float* bvec  = (const float*)d_in[2];   // (8,)
  const float* scale = (const float*)d_in[3];   // (8,)
  const float* bias  = (const float*)d_in[4];   // (8,)
  const float* qw    = (const float*)d_in[5];   // (3, 8, 3)
  float* out = (float*)d_out;                   // (8192, 8)

  // Intermediate angles buffer: prefer workspace; d_out has identical shape
  // and kernel 2 reads its 8 angles before overwriting them, so it is a safe
  // fallback.
  float* angles = (ws_size >= (size_t)BATCH * NQ * sizeof(float))
                      ? (float*)d_ws : out;

  // 512 M-tiles, 8 waves (tiles) per 256-thread block -> 64 blocks
  qnn_gemm_angles<<<64, 256, 0, stream>>>(x, W, bvec, scale, bias, angles);
  // 8192 batch elements, 1 per wave, 8 waves per block -> 1024 blocks
  qnn_sim<<<BATCH / 8, 256, 0, stream>>>(angles, qw, out);
}
